// TransformerEncoder_16226386444358
// MI455X (gfx1250) — matrix operand order repaired and checked by reading, not compiled, not run on hardware
//
#include <hip/hip_runtime.h>
#include <hip/hip_bf16.h>
#include <math.h>

typedef __attribute__((ext_vector_type(16))) _Float16 v16h;
typedef __attribute__((ext_vector_type(8)))  _Float16 h8;
typedef __attribute__((ext_vector_type(8)))  float    v8f;

#define NLAYER 4
#define DMODEL 1024
#define NHEAD  8
#define HDIM   128
#define FFN    4096
#define BB     8
#define LL     512
#define LP2    (2*LL)          // 1024 relative positions
#define MROWS  (BB*LL)         // 4096

// GEMM epilogue flags
#define GF_BIAS   1
#define GF_RELU   2
#define GF_OUTF32 4
#define GF_OUTF16 8

struct GemmArgs {
    const _Float16* A;    // M x K row-major (lda)
    const _Float16* Bt;   // N x K row-major (ldb)  == B transposed
    float*          C;    // M x N f32 (ldc)
    _Float16*       C16;  // M x N f16 (ldc16)
    const float*    bias; // per-column
    int M, N, K, lda, ldb, ldc, ldc16, flags, batchInner;
    long long aOuter, aInner, bOuter, bInner, cOuter, cInner, c16Outer, c16Inner;
};

// ---------------------------------------------------------------------------
// WMMA GEMM: 256 threads = 8 waves, block tile 128x128, K-step 32.
// LDS holds tiles in fragment-major order so each lane reads its v16h
// fragment as one contiguous 32B ds_load_b128 pair. Wave grid 4(M)x2(N);
// each wave owns a 32x64 subtile = 2x4 v_wmma_f32_16x16x32_f16 accumulators.
// Global pointers are hoisted and bumped by +64B/iter; tile k+1 is loaded
// into registers before the barrier so vmem overlaps the WMMAs of tile k,
// and the next tile is prefetched unconditionally (speculative, safe OOB).
// ---------------------------------------------------------------------------
__global__ __launch_bounds__(256) void k_gemm_wmma(GemmArgs p) {
    __shared__ _Float16 ldsA[8 * 512];   // 8 m-frags * 32 lanes * 16 halves
    __shared__ _Float16 ldsB[8 * 512];   // 8 n-frags

    const int tid  = threadIdx.x;
    const int lane = tid & 31;
    const int wid  = tid >> 5;
    const int wm   = wid & 3;            // 32-row slab
    const int wn   = wid >> 2;           // 64-col slab
    const int m0   = blockIdx.y * 128;
    const int n0   = blockIdx.x * 128;

    const long long z  = blockIdx.z;
    const long long zo = z / p.batchInner;
    const long long zi = z % p.batchInner;
    const _Float16* __restrict__ A  = p.A  + zo * p.aOuter + zi * p.aInner;
    const _Float16* __restrict__ Bt = p.Bt + zo * p.bOuter + zi * p.bInner;

    // per-thread staging assignment: two 16B groups for A, two for B
    const int row0 = tid >> 2;           // 0..63
    const int row1 = row0 + 64;          // 64..127
    const int kg   = tid & 3;            // 8-half K group within K-step
    const _Float16* __restrict__ pA0 = A  + (size_t)(m0 + row0) * p.lda + kg * 8;
    const _Float16* __restrict__ pA1 = A  + (size_t)(m0 + row1) * p.lda + kg * 8;
    const _Float16* __restrict__ pB0 = Bt + (size_t)(n0 + row0) * p.ldb + kg * 8;
    const _Float16* __restrict__ pB1 = Bt + (size_t)(n0 + row1) * p.ldb + kg * 8;
    // LDS destinations (loop-invariant); row1 & 15 == row0 & 15
    const int laneA = ((row0 & 15) + ((kg & 1) << 4)) * 16 + ((kg >> 1) << 3);
    const int laneB = ((row0 & 15) + ((kg & 1) << 4)) * 16 + ((kg >> 1) << 3);
    const int da0 = (row0 >> 4) * 512 + laneA;
    const int da1 = (row1 >> 4) * 512 + laneA;
    const int db0 = (row0 >> 4) * 512 + laneB;
    const int db1 = (row1 >> 4) * 512 + laneB;

    v8f acc[2][4] = {};

    const int nk = p.K >> 5;
    for (int kt = 0; kt < nk; ++kt) {
        // issue global loads for this tile before waiting on the barrier so
        // they overlap the previous iteration's WMMAs
        h8 a0 = *(const h8*)pA0;
        h8 a1 = *(const h8*)pA1;
        h8 b0 = *(const h8*)pB0;
        h8 b1 = *(const h8*)pB1;
        __builtin_prefetch((const void*)(pA0 + 32), 0, 0);   // global_prefetch_b8
        __builtin_prefetch((const void*)(pB0 + 32), 0, 0);
        pA0 += 32; pA1 += 32; pB0 += 32; pB1 += 32;

        __syncthreads();                 // previous compute phase done
        *(h8*)&ldsA[da0] = a0;
        *(h8*)&ldsA[da1] = a1;
        *(h8*)&ldsB[db0] = b0;
        *(h8*)&ldsB[db1] = b1;
        __syncthreads();

        v16h af[2], bf[4];
        #pragma unroll
        for (int i = 0; i < 2; ++i)
            af[i] = *(const v16h*)&ldsA[(wm * 2 + i) * 512 + lane * 16];
        #pragma unroll
        for (int j = 0; j < 4; ++j)
            bf[j] = *(const v16h*)&ldsB[(wn * 4 + j) * 512 + lane * 16];
        #pragma unroll
        for (int i = 0; i < 2; ++i)
            #pragma unroll
            for (int j = 0; j < 4; ++j)
                acc[i][j] = __builtin_amdgcn_wmma_f32_16x16x32_f16(
                    false, af[i], false, bf[j], (short)0, acc[i][j], false, false);
    }

    // epilogue: C/D frag: VGPR r -> row r + 8*(lane>>4), col = lane&15
    const int colL  = lane & 15;
    const int rbase = (lane >> 4) * 8;
    float*     C   = p.C   ? (p.C   + zo * p.cOuter   + zi * p.cInner)   : nullptr;
    _Float16*  C16 = p.C16 ? (p.C16 + zo * p.c16Outer + zi * p.c16Inner) : nullptr;
    #pragma unroll
    for (int i = 0; i < 2; ++i) {
        const int r0 = m0 + wm * 32 + i * 16 + rbase;
        #pragma unroll
        for (int j = 0; j < 4; ++j) {
            const int c = n0 + wn * 64 + j * 16 + colL;
            const float bval = (p.flags & GF_BIAS) ? p.bias[c] : 0.0f;
            #pragma unroll
            for (int r = 0; r < 8; ++r) {
                float v = acc[i][j][r] + bval;
                if (p.flags & GF_RELU) v = fmaxf(v, 0.0f);
                if (p.flags & GF_OUTF32) C[(size_t)(r0 + r) * p.ldc + c] = v;
                if (p.flags & GF_OUTF16) C16[(size_t)(r0 + r) * p.ldc16 + c] = (_Float16)v;
            }
        }
    }
}

// ---------------------------------------------------------------------------
// small helper kernels
// ---------------------------------------------------------------------------
__global__ void k_f32_to_f16(const float* in, _Float16* out, size_t n) {
    size_t i = (size_t)blockIdx.x * 256 + threadIdx.x;
    if (i < n) out[i] = (_Float16)in[i];
}

// W (K x N f32, row-major) -> Wt (N x K f16, row-major)
__global__ void k_wtranspose(const float* W, _Float16* Wt, int K, int N) {
    size_t id = (size_t)blockIdx.x * 256 + threadIdx.x;
    if (id >= (size_t)K * N) return;
    int n = (int)(id % N), k = (int)(id / N);
    Wt[(size_t)n * K + k] = (_Float16)W[id];
}

// sinusoidal relative position table: pos[t][j], t in [0,1024) ~ rel pos t-512
__global__ void k_pos(float* posf, _Float16* pos16) {
    size_t id = (size_t)blockIdx.x * 256 + threadIdx.x;   // 1024*128
    if (id >= (size_t)LP2 * HDIM) return;
    int t = (int)(id / HDIM), j = (int)(id % HDIM);
    int jj = (j < 64) ? j : j - 64;
    float inv = __expf(-(float)jj * (logf(10000.0f) / 63.0f));
    float ang = (float)(t - LL) * inv;
    float v   = (j < 64) ? __sinf(ang) : __cosf(ang);
    posf[id]  = v;
    pos16[id] = (_Float16)v;
}

// Dn[n][t] = sum_d r_w[n][d] * pos[t][d]   (8 x 1024, K=128)
__global__ void k_dpos(const float* rw, const float* posf, float* Dn) {
    int id = blockIdx.x * 256 + threadIdx.x;              // 8192
    if (id >= NHEAD * LP2) return;
    int n = id / LP2, t = id % LP2;
    float s = 0.f;
    for (int d = 0; d < HDIM; ++d) s += rw[n * HDIM + d] * posf[t * HDIM + d];
    Dn[id] = s;
}

// split qkv (f16, rows (b,l) x 3072) -> per-head q, q+r_r, k, and v transposed
__global__ void k_split_heads(const _Float16* qkv, const float* rr,
                              _Float16* qh, _Float16* rwq, _Float16* kh, _Float16* vT) {
    size_t id = (size_t)blockIdx.x * 256 + threadIdx.x;   // 64*512*128
    if (id >= (size_t)BB * NHEAD * LL * HDIM) return;
    int d  = (int)(id & (HDIM - 1));
    int l  = (int)((id >> 7) & (LL - 1));
    int bn = (int)(id >> 16);
    int n  = bn & (NHEAD - 1), b = bn >> 3;
    size_t src = ((size_t)(b * LL + l)) * (3 * DMODEL) + (size_t)n * HDIM + d;
    _Float16 qv = qkv[src];
    _Float16 kv = qkv[src + DMODEL];
    _Float16 vv = qkv[src + 2 * DMODEL];
    size_t hd = ((size_t)bn * LL + l) * HDIM + d;
    qh[hd]  = qv;
    rwq[hd] = (_Float16)((float)qv + rr[n * HDIM + d]);
    kh[hd]  = kv;
    vT[((size_t)bn * HDIM + d) * LL + l] = vv;
}

// fused rel-shift gather + mask + softmax; one 256-thread block per (b,n,q) row
// score[k] = AC[q,k] + Bq[q, L+k-q] + Dn[n, L+k-q] + Ek[k, L+q-k]
__global__ __launch_bounds__(256) void k_softmax(const float* AC, const _Float16* Bq,
        const _Float16* Ek, const float* Dn, const int* mask, _Float16* attnP) {
    __shared__ float red[256];
    const int tid = threadIdx.x;
    const int z   = blockIdx.x;          // bn*512 + q
    const int q   = z & (LL - 1);
    const int bn  = z >> 9;
    const int n   = bn & (NHEAD - 1);
    const int b   = bn >> 3;
    const float*    acr = AC + ((size_t)bn * LL + q) * LL;
    const _Float16* bqr = Bq + ((size_t)bn * LL + q) * LP2;
    const _Float16* ekb = Ek + (size_t)bn * LL * LP2;
    const float*    dnr = Dn + (size_t)n * LP2;
    const int*      mr  = mask + b * LL;
    float s[2];
    #pragma unroll
    for (int t = 0; t < 2; ++t) {
        int k  = tid + t * 256;
        int t1 = LL + k - q;
        int t2 = LL + q - k;
        float v = acr[k] + (float)bqr[t1] + dnr[t1] + (float)ekb[(size_t)k * LP2 + t2];
        if (mr[k] == 0) v = -INFINITY;
        s[t] = v;
    }
    float m = fmaxf(s[0], s[1]);
    red[tid] = m; __syncthreads();
    for (int st = 128; st > 0; st >>= 1) {
        if (tid < st) red[tid] = fmaxf(red[tid], red[tid + st]);
        __syncthreads();
    }
    m = red[0]; __syncthreads();
    float e0 = __expf(s[0] - m), e1 = __expf(s[1] - m);
    red[tid] = e0 + e1; __syncthreads();
    for (int st = 128; st > 0; st >>= 1) {
        if (tid < st) red[tid] += red[tid + st];
        __syncthreads();
    }
    float inv = 1.0f / red[0];
    _Float16* outr = attnP + ((size_t)bn * LL + q) * LL;
    outr[tid]       = (_Float16)(e0 * inv);
    outr[tid + 256] = (_Float16)(e1 * inv);
}

// x = LayerNorm(base + delta); writes f32 (d_out chain) and f16 (next GEMM A)
__global__ __launch_bounds__(256) void k_add_ln(const float* base, const float* delta,
        const float* g, const float* bta, float* outF, _Float16* out16) {
    __shared__ float red[256];
    const int row = blockIdx.x, tid = threadIdx.x;
    const float* xr = base  + (size_t)row * DMODEL;
    const float* dr = delta + (size_t)row * DMODEL;
    float v[4]; float s = 0.f;
    #pragma unroll
    for (int t = 0; t < 4; ++t) { v[t] = xr[tid + t * 256] + dr[tid + t * 256]; s += v[t]; }
    red[tid] = s; __syncthreads();
    for (int st = 128; st > 0; st >>= 1) { if (tid < st) red[tid] += red[tid + st]; __syncthreads(); }
    const float mu = red[0] * (1.0f / DMODEL); __syncthreads();
    float s2 = 0.f;
    #pragma unroll
    for (int t = 0; t < 4; ++t) { float d = v[t] - mu; s2 += d * d; }
    red[tid] = s2; __syncthreads();
    for (int st = 128; st > 0; st >>= 1) { if (tid < st) red[tid] += red[tid + st]; __syncthreads(); }
    const float rstd = rsqrtf(red[0] * (1.0f / DMODEL) + 1e-5f);
    #pragma unroll
    for (int t = 0; t < 4; ++t) {
        int c = tid + t * 256;
        float o = (v[t] - mu) * rstd * g[c] + bta[c];
        outF[(size_t)row * DMODEL + c]  = o;
        out16[(size_t)row * DMODEL + c] = (_Float16)o;
    }
}

// ---------------------------------------------------------------------------
// orchestration
// ---------------------------------------------------------------------------
static inline void launch_gemm(GemmArgs a, int batches, hipStream_t s) {
    dim3 grid(a.N / 128, a.M / 128, batches);
    hipLaunchKernelGGL(k_gemm_wmma, grid, dim3(256), 0, s, a);
}

extern "C" void kernel_launch(void* const* d_in, const int* in_sizes, int n_in,
                              void* d_out, int out_size, void* d_ws, size_t ws_size,
                              hipStream_t stream) {
    (void)in_sizes; (void)n_in; (void)out_size; (void)ws_size;
    const float* x_in  = (const float*)d_in[0];
    const int*   mask  = (const int*)d_in[1];
    const float* qkv_w = (const float*)d_in[2];
    const float* r_r   = (const float*)d_in[3];
    const float* r_w   = (const float*)d_in[4];
    const float* ln1g  = (const float*)d_in[5];
    const float* ln1b  = (const float*)d_in[6];
    const float* ln2g  = (const float*)d_in[7];
    const float* ln2b  = (const float*)d_in[8];
    const float* w1    = (const float*)d_in[9];
    const float* b1    = (const float*)d_in[10];
    const float* w2    = (const float*)d_in[11];
    const float* b2    = (const float*)d_in[12];
    float* xout = (float*)d_out;

    char* ws = (char*)d_ws;
    const size_t MB = 1ull << 20;
    const size_t OFF_X16  = 0;               // 8  MiB: x f16 (4096x1024)
    const size_t OFF_WT   = OFF_X16 + 8*MB;  // 8  MiB: transposed weights f16
    const size_t OFF_BIG  = OFF_WT  + 8*MB;  // 32 MiB: qkv16 / h16 / attnP
    const size_t OFF_QH   = OFF_BIG + 32*MB; // 8  MiB
    const size_t OFF_RWQ  = OFF_QH  + 8*MB;  // 8  MiB
    const size_t OFF_KH   = OFF_RWQ + 8*MB;  // 8  MiB
    const size_t OFF_VT   = OFF_KH  + 8*MB;  // 8  MiB
    const size_t OFF_AC   = OFF_VT  + 8*MB;  // 64 MiB: AC f32 / y f32
    const size_t OFF_BQ   = OFF_AC  + 64*MB; // 64 MiB: Bq f16 / ao f32
    const size_t OFF_EK   = OFF_BQ  + 64*MB; // 64 MiB: Ek f16
    const size_t OFF_POSF = OFF_EK  + 64*MB; // 512 KiB
    const size_t OFF_P16  = OFF_POSF + 1*MB; // 256 KiB
    const size_t OFF_DN   = OFF_P16  + 1*MB; // 32 KiB

    _Float16* X16   = (_Float16*)(ws + OFF_X16);
    _Float16* WT    = (_Float16*)(ws + OFF_WT);
    _Float16* BIG   = (_Float16*)(ws + OFF_BIG);   // qkv16 -> attnP -> h16
    _Float16* QH    = (_Float16*)(ws + OFF_QH);
    _Float16* RWQ   = (_Float16*)(ws + OFF_RWQ);
    _Float16* KH    = (_Float16*)(ws + OFF_KH);
    _Float16* VT    = (_Float16*)(ws + OFF_VT);
    float*    ACf   = (float*)   (ws + OFF_AC);
    _Float16* BQ    = (_Float16*)(ws + OFF_BQ);
    float*    AOf   = (float*)   (ws + OFF_BQ);    // reuses BQ region
    _Float16* EK    = (_Float16*)(ws + OFF_EK);
    float*    POSF  = (float*)   (ws + OFF_POSF);
    _Float16* POS16 = (_Float16*)(ws + OFF_P16);
    float*    DN    = (float*)   (ws + OFF_DN);

    hipLaunchKernelGGL(k_pos, dim3((LP2*HDIM)/256), dim3(256), 0, stream, POSF, POS16);
    hipLaunchKernelGGL(k_f32_to_f16, dim3((MROWS*DMODEL)/256), dim3(256), 0, stream,
                       x_in, X16, (size_t)MROWS * DMODEL);

    for (int layer = 0; layer < NLAYER; ++layer) {
        const float* base = (layer == 0) ? x_in : xout;

        // ---- QKV projection: qkv16 = x @ Wqkv (f16 out) ----
        hipLaunchKernelGGL(k_wtranspose, dim3((DMODEL*3*DMODEL)/256), dim3(256), 0, stream,
                           qkv_w + (size_t)layer * DMODEL * 3 * DMODEL, WT, DMODEL, 3 * DMODEL);
        GemmArgs g{};
        g.A = X16; g.Bt = WT; g.C16 = BIG;
        g.M = MROWS; g.N = 3 * DMODEL; g.K = DMODEL;
        g.lda = DMODEL; g.ldb = DMODEL; g.ldc16 = 3 * DMODEL;
        g.flags = GF_OUTF16; g.batchInner = 1;
        launch_gemm(g, 1, stream);

        // ---- split heads, add r_r bias, transpose v ----
        hipLaunchKernelGGL(k_split_heads, dim3((BB*NHEAD*LL*HDIM)/256), dim3(256), 0, stream,
                           BIG, r_r + (size_t)layer * NHEAD * HDIM, QH, RWQ, KH, VT);

        // ---- AC = (q + r_r) @ k^T  (batched 64, f32) ----
        g = GemmArgs{};
        g.A = RWQ; g.Bt = KH; g.C = ACf;
        g.M = LL; g.N = LL; g.K = HDIM;
        g.lda = HDIM; g.ldb = HDIM; g.ldc = LL;
        g.aOuter = (long long)LL * HDIM; g.bOuter = (long long)LL * HDIM;
        g.cOuter = (long long)LL * LL;
        g.flags = GF_OUTF32; g.batchInner = 1;
        launch_gemm(g, BB * NHEAD, stream);

        // ---- Bq = q @ pos^T, Ek = k @ pos^T  (batched 64, f16 out) ----
        g = GemmArgs{};
        g.A = QH; g.Bt = POS16; g.C16 = BQ;
        g.M = LL; g.N = LP2; g.K = HDIM;
        g.lda = HDIM; g.ldb = HDIM; g.ldc16 = LP2;
        g.aOuter = (long long)LL * HDIM; g.c16Outer = (long long)LL * LP2;
        g.flags = GF_OUTF16; g.batchInner = 1;
        launch_gemm(g, BB * NHEAD, stream);
        g.A = KH; g.C16 = EK;
        launch_gemm(g, BB * NHEAD, stream);

        // ---- Dn = r_w @ pos^T (tiny) ----
        hipLaunchKernelGGL(k_dpos, dim3((NHEAD*LP2)/256), dim3(256), 0, stream,
                           r_w + (size_t)layer * NHEAD * HDIM, POSF, DN);

        // ---- fused shift-gather + softmax -> attnP (f16, reuses BIG) ----
        hipLaunchKernelGGL(k_softmax, dim3(BB*NHEAD*LL), dim3(256), 0, stream,
                           ACf, BQ, EK, DN, mask, BIG);

        // ---- attention output: ao = attn @ v  (head-merged into (b,l,d)) ----
        g = GemmArgs{};
        g.A = BIG; g.Bt = VT; g.C = AOf;
        g.M = LL; g.N = HDIM; g.K = LL;
        g.lda = LL; g.ldb = LL; g.ldc = DMODEL;
        g.batchInner = NHEAD;
        g.aInner = (long long)LL * LL;  g.aOuter = (long long)NHEAD * LL * LL;
        g.bInner = (long long)HDIM * LL; g.bOuter = (long long)NHEAD * HDIM * LL;
        g.cInner = HDIM;                g.cOuter = (long long)LL * DMODEL;
        g.flags = GF_OUTF32;
        launch_gemm(g, BB * NHEAD, stream);

        // ---- x = LN(x + ao) ----
        hipLaunchKernelGGL(k_add_ln, dim3(MROWS), dim3(256), 0, stream,
                           base, AOf, ln1g + (size_t)layer * DMODEL, ln1b + (size_t)layer * DMODEL,
                           xout, X16);

        // ---- FFN1: h16 = relu(x @ w1 + b1) ----
        hipLaunchKernelGGL(k_wtranspose, dim3((DMODEL*FFN)/256), dim3(256), 0, stream,
                           w1 + (size_t)layer * DMODEL * FFN, WT, DMODEL, FFN);
        g = GemmArgs{};
        g.A = X16; g.Bt = WT; g.C16 = BIG; g.bias = b1 + (size_t)layer * FFN;
        g.M = MROWS; g.N = FFN; g.K = DMODEL;
        g.lda = DMODEL; g.ldb = DMODEL; g.ldc16 = FFN;
        g.flags = GF_BIAS | GF_RELU | GF_OUTF16; g.batchInner = 1;
        launch_gemm(g, 1, stream);

        // ---- FFN2: y = h @ w2 + b2 (f32, reuses AC region) ----
        hipLaunchKernelGGL(k_wtranspose, dim3((FFN*DMODEL)/256), dim3(256), 0, stream,
                           w2 + (size_t)layer * FFN * DMODEL, WT, FFN, DMODEL);
        g = GemmArgs{};
        g.A = BIG; g.Bt = WT; g.C = ACf; g.bias = b2 + (size_t)layer * DMODEL;
        g.M = MROWS; g.N = DMODEL; g.K = FFN;
        g.lda = FFN; g.ldb = FFN; g.ldc = DMODEL;
        g.flags = GF_BIAS | GF_OUTF32; g.batchInner = 1;
        launch_gemm(g, 1, stream);

        // ---- x = LN(x + y) ----
        hipLaunchKernelGGL(k_add_ln, dim3(MROWS), dim3(256), 0, stream,
                           xout, ACf, ln2g + (size_t)layer * DMODEL, ln2b + (size_t)layer * DMODEL,
                           xout, X16);
    }
}
